// GODE_74217034875616
// MI455X (gfx1250) — matrix operand order repaired and checked by reading, not compiled, not run on hardware
//
#include <hip/hip_runtime.h>
#include <hip/hip_bf16.h>
#include <math.h>

typedef __attribute__((ext_vector_type(16))) __bf16 v16bf;
typedef __attribute__((ext_vector_type(8)))  float  v8f;

#define MTOT 32768   // B*N rows
#define KDIM 1024    // 2H
#define HDIM 512     // H
#define ODE_DT 0.1f

#define LDSPAD 8     // pad to 72 ushorts/row = 144B (16B aligned, bank-spread)

__device__ __forceinline__ unsigned short f2bf(float f) {
    unsigned u = __float_as_uint(f);
    u += 0x7fffu + ((u >> 16) & 1u);      // round-to-nearest-even
    return (unsigned short)(u >> 16);
}
__device__ __forceinline__ float bf2f(unsigned short h) {
    return __uint_as_float(((unsigned)h) << 16);
}
__device__ __forceinline__ unsigned pack2(float a, float b) {
    return (unsigned)f2bf(a) | ((unsigned)f2bf(b) << 16);
}

// W [K][Nw] fp32 -> Wt [Nw][K] bf16 (write-coalesced)
__global__ void wtrans_bf16(const float* __restrict__ W,
                            unsigned short* __restrict__ Wt, int K, int Nw) {
    int idx = blockIdx.x * blockDim.x + threadIdx.x;
    if (idx >= K * Nw) return;
    int n = idx / K, k = idx - n * K;
    Wt[idx] = f2bf(W[k * Nw + n]);
}

// Workgroup tile: M=256, N=128, K-step 64. 8 waves, each owns 64x64 (4x4 WMMA tiles).
// EP 0: Y = bf16( Xres + 0.1*tanh(acc+bias) )   (ODE euler step, Nw==KDIM)
// EP 1: Y = bf16( relu(acc+bias) )              (Nw==KDIM)
// EP 2: Yf = tanh(acc+bias)  fp32               (final, Nw==HDIM)
// CONCAT_A: A operand (and EP0 residual) read as fp32 concat(inp,hz), converted on the fly.
template <int EP, bool CONCAT_A>
__global__ __launch_bounds__(256)
void gemm_bf16_wmma(const unsigned short* __restrict__ A,    // [MTOT][KDIM] bf16 (if !CONCAT_A)
                    const float* __restrict__ inpA,          // [MTOT][HDIM] fp32 (if CONCAT_A)
                    const float* __restrict__ hzA,           // [MTOT][HDIM] fp32 (if CONCAT_A)
                    const unsigned short* __restrict__ Wt,   // [Nw][KDIM]  bf16 (transposed)
                    const float* __restrict__ bias,          // [Nw]
                    const unsigned short* __restrict__ Xres, // residual bf16 (EP0, !CONCAT_A)
                    unsigned short* __restrict__ Ybf,
                    float* __restrict__ Yf,
                    int Nw) {
    __shared__ unsigned short ldsA[256][64 + LDSPAD];
    __shared__ unsigned short ldsB[128][64 + LDSPAD];

    const int tid  = threadIdx.x;
    const int lane = tid & 31;
    const int wv   = tid >> 5;          // 8 waves: 4 in M x 2 in N
    const int wm   = (wv & 3) << 6;     // 0/64/128/192
    const int wn   = (wv >> 2) << 6;    // 0/64
    const int l16  = lane & 15;
    const int hi   = lane >> 4;         // lane half (ISA A/B layout split)

    const int n0 = blockIdx.x * 128;
    const int m0 = blockIdx.y * 256;

    v8f acc[4][4];
    const v8f vzero = {0.f, 0.f, 0.f, 0.f, 0.f, 0.f, 0.f, 0.f};
#pragma unroll
    for (int i = 0; i < 4; ++i)
#pragma unroll
        for (int j = 0; j < 4; ++j) acc[i][j] = vzero;

    for (int k0 = 0; k0 < KDIM; k0 += 64) {
        __syncthreads();   // previous iteration's fragment reads done before overwrite
        // ---- Stage A tile: 256 rows x 64 cols bf16 = 2048 8-elem chunks, 8/thread
#pragma unroll
        for (int it = 0; it < 8; ++it) {
            int i = tid + it * 256;
            int r = i >> 3, q = i & 7;
            if (CONCAT_A) {
                int c = k0 + q * 8;   // 512 boundary is 8-aligned: chunk is all-inp or all-hz
                const float* src = (c < HDIM)
                    ? (inpA + (size_t)(m0 + r) * HDIM + c)
                    : (hzA  + (size_t)(m0 + r) * HDIM + (c - HDIM));
                float4 f0 = *(const float4*)src;
                float4 f1 = *(const float4*)(src + 4);
                uint4 o;
                o.x = pack2(f0.x, f0.y); o.y = pack2(f0.z, f0.w);
                o.z = pack2(f1.x, f1.y); o.w = pack2(f1.z, f1.w);
                *(uint4*)&ldsA[r][q * 8] = o;
            } else {
                *(uint4*)&ldsA[r][q * 8] =
                    *(const uint4*)&A[(size_t)(m0 + r) * KDIM + k0 + q * 8];
            }
        }
        // ---- Stage B tile: 128 rows x 64 cols bf16 = 1024 chunks, 4/thread
#pragma unroll
        for (int it = 0; it < 4; ++it) {
            int i = tid + it * 256;
            int r = i >> 3, q = i & 7;
            *(uint4*)&ldsB[r][q * 8] =
                *(const uint4*)&Wt[(size_t)(n0 + r) * KDIM + k0 + q * 8];
        }
        __syncthreads();

        // ---- Two 32-K micro-steps per LDS tile
#pragma unroll
        for (int ks = 0; ks < 2; ++ks) {
            const int koff = ks * 32;
            // A frags: lanes 0-15 -> K {0..7,16..23}; lanes 16-31 -> K {8..15,24..31}
            v16bf afrag[4];
#pragma unroll
            for (int mt = 0; mt < 4; ++mt) {
                const unsigned short* p = &ldsA[wm + mt * 16 + l16][koff + hi * 8];
                ((uint4*)&afrag[mt])[0] = *(const uint4*)p;
                ((uint4*)&afrag[mt])[1] = *(const uint4*)(p + 16);
            }
            // B frags: column = l16; lanes 0-15 -> K {0..7,16..23}; lanes 16-31 -> K {8..15,24..31}
            v16bf bfrag[4];
#pragma unroll
            for (int nt = 0; nt < 4; ++nt) {
                const unsigned short* p = &ldsB[wn + nt * 16 + l16][koff + hi * 8];
                ((uint4*)&bfrag[nt])[0] = *(const uint4*)p;
                ((uint4*)&bfrag[nt])[1] = *(const uint4*)(p + 16);
            }
#pragma unroll
            for (int mt = 0; mt < 4; ++mt)
#pragma unroll
                for (int nt = 0; nt < 4; ++nt)
                    acc[mt][nt] = __builtin_amdgcn_wmma_f32_16x16x32_bf16(
                        false, afrag[mt], false, bfrag[nt], (short)0, acc[mt][nt],
                        false, false);
        }
    }

    // ---- Epilogue: C/D layout -> row = hi*8 + e, col = l16
#pragma unroll
    for (int mt = 0; mt < 4; ++mt) {
#pragma unroll
        for (int nt = 0; nt < 4; ++nt) {
            const int col = n0 + wn + nt * 16 + l16;
            const float bv = bias[col];
#pragma unroll
            for (int e = 0; e < 8; ++e) {
                const int row = m0 + wm + mt * 16 + hi * 8 + e;
                float v = acc[mt][nt][e] + bv;
                if (EP == 0) {
                    float x;
                    if (CONCAT_A) {
                        x = (col < HDIM)
                            ? inpA[(size_t)row * HDIM + col]
                            : hzA[(size_t)row * HDIM + (col - HDIM)];
                    } else {
                        x = bf2f(Xres[(size_t)row * KDIM + col]);
                    }
                    Ybf[(size_t)row * KDIM + col] = f2bf(x + ODE_DT * tanhf(v));
                } else if (EP == 1) {
                    Ybf[(size_t)row * KDIM + col] = f2bf(fmaxf(v, 0.f));
                } else {
                    Yf[(size_t)row * (size_t)Nw + col] = tanhf(v);
                }
            }
        }
    }
}

extern "C" void kernel_launch(void* const* d_in, const int* in_sizes, int n_in,
                              void* d_out, int out_size, void* d_ws, size_t ws_size,
                              hipStream_t stream) {
    (void)in_sizes; (void)n_in; (void)out_size; (void)ws_size;
    const float* inp = (const float*)d_in[0];
    const float* hz  = (const float*)d_in[1];
    const float* Wf  = (const float*)d_in[2];
    const float* bfv = (const float*)d_in[3];
    const float* W1  = (const float*)d_in[4];
    const float* b1  = (const float*)d_in[5];
    const float* W2  = (const float*)d_in[6];
    const float* b2  = (const float*)d_in[7];
    float* out = (float*)d_out;

    char* ws = (char*)d_ws;
    unsigned short* bufA = (unsigned short*)ws;                              // 64 MiB
    unsigned short* bufB = (unsigned short*)(ws + (size_t)MTOT * KDIM * 2);  // 64 MiB
    unsigned short* WfT  = (unsigned short*)(ws + (size_t)MTOT * KDIM * 4);
    unsigned short* W1T  = WfT + (size_t)KDIM * KDIM;
    unsigned short* W2T  = W1T + (size_t)KDIM * KDIM;

    // Weight convert + transpose (tiny)
    wtrans_bf16<<<(KDIM * KDIM + 255) / 256, 256, 0, stream>>>(Wf, WfT, KDIM, KDIM);
    wtrans_bf16<<<(KDIM * KDIM + 255) / 256, 256, 0, stream>>>(W1, W1T, KDIM, KDIM);
    wtrans_bf16<<<(KDIM * HDIM + 255) / 256, 256, 0, stream>>>(W2, W2T, KDIM, HDIM);

    dim3 blk(256);
    dim3 g1(KDIM / 128, MTOT / 256);
    // Euler step 1 (A = concat(inp,hz) fp32, fused convert): H1 -> bufB
    gemm_bf16_wmma<0, true><<<g1, blk, 0, stream>>>(
        nullptr, inp, hz, WfT, bfv, nullptr, bufB, nullptr, KDIM);
    // Euler step 2: H2 = H1 + 0.1*tanh(H1 @ Wf + bf) -> bufA
    gemm_bf16_wmma<0, false><<<g1, blk, 0, stream>>>(
        bufB, nullptr, nullptr, WfT, bfv, bufB, bufA, nullptr, KDIM);
    // H3 = relu(H2 @ W1 + b1) -> bufB
    gemm_bf16_wmma<1, false><<<g1, blk, 0, stream>>>(
        bufA, nullptr, nullptr, W1T, b1, nullptr, bufB, nullptr, KDIM);
    // out = tanh(H3 @ W2 + b2)  (fp32)
    dim3 g2(HDIM / 128, MTOT / 256);
    gemm_bf16_wmma<2, false><<<g2, blk, 0, stream>>>(
        bufB, nullptr, nullptr, W2T, b2, nullptr, nullptr, out, HDIM);
}
